// naive_gat_like_55757265437271
// MI455X (gfx1250) — matrix operand order repaired and checked by reading, not compiled, not run on hardware
//
#include <hip/hip_runtime.h>
#include <hip/hip_bf16.h>

// ---------------------------------------------------------------------------
// GAT-like attention for MI455X (gfx1250, wave32, WMMA).
//   s = x@(w_w^T a1) + w_b.a1   d = x@(w_w^T a2) + w_b.a2     (DK folded away)
//   Dmax = max_j d_j ;  m_i = leaky(s_i + Dmax)   (exact row max, monotone)
//   w_ij = exp(leaky(s_i+d_j) - m_i)
//        = (s_i+d_j>=0) ? P_i*E_j : Q_i*F_j      (all factors <= 1, no exp in
//          the hot loop -> no TRANS-pipe contention with 16-bit WMMA)
//   out[i] = (sum_j w_ij * v[j]) / Z_i,  Z via WMMA against a ones matrix
//            (bit-consistent with the bf16 numerator, lands in D-layout).
// Big GEMM (69 GFLOP) runs on v_wmma_f32_16x16x32_bf16 with f32 accumulation.
// ---------------------------------------------------------------------------

typedef __attribute__((ext_vector_type(16))) __bf16 v16bf;
typedef __attribute__((ext_vector_type(8)))  __bf16 v8bf;
typedef __attribute__((ext_vector_type(8)))  float  v8f;

#define NROWS 8192
#define KDIM  512
#define DKP   128
#define DVC   512

__device__ __forceinline__ float leaky01(float t) {
    return t >= 0.0f ? t : 0.01f * t;
}

// ---- kernel 1: u1[c] = sum_k w_w[k][c]*a[k]; u2 with a[DK+k]; bias scalars ----
__global__ void prep_u_kernel(const float* __restrict__ w_w,
                              const float* __restrict__ w_b,
                              const float* __restrict__ a,
                              float* __restrict__ u1, float* __restrict__ u2,
                              float* __restrict__ cpair) {
    int c = threadIdx.x;
    if (c < KDIM) {
        float s1 = 0.0f, s2 = 0.0f;
        for (int k = 0; k < DKP; ++k) {
            float w = w_w[k * KDIM + c];
            s1 += w * a[k];
            s2 += w * a[DKP + k];
        }
        u1[c] = s1;
        u2[c] = s2;
    }
    if (threadIdx.x == 0) {
        float c1 = 0.0f, c2 = 0.0f;
        for (int k = 0; k < DKP; ++k) {
            c1 += w_b[k] * a[k];
            c2 += w_b[k] * a[DKP + k];
        }
        cpair[0] = c1;
        cpair[1] = c2;
    }
}

// ---- kernel 2: s[i] = x[i].u1 + c1 ; d[i] = x[i].u2 + c2 (one wave per row) ----
__global__ void compute_sd_kernel(const float* __restrict__ x,
                                  const float* __restrict__ u1,
                                  const float* __restrict__ u2,
                                  const float* __restrict__ cpair,
                                  float* __restrict__ s, float* __restrict__ d) {
    int lane = threadIdx.x & 31;
    int wave = threadIdx.x >> 5;
    int row  = blockIdx.x * 8 + wave;
    const float* xr = x + (size_t)row * KDIM;
    float a1 = 0.0f, a2 = 0.0f;
    for (int c = lane; c < KDIM; c += 32) {
        float xv = xr[c];
        a1 += xv * u1[c];
        a2 += xv * u2[c];
    }
    #pragma unroll
    for (int off = 16; off > 0; off >>= 1) {
        a1 += __shfl_xor(a1, off, 32);
        a2 += __shfl_xor(a2, off, 32);
    }
    if (lane == 0) {
        s[row] = a1 + cpair[0];
        d[row] = a2 + cpair[1];
    }
}

// ---- kernel 3: dmax = max_j d[j] (single block) ----
__global__ void reduce_dmax_kernel(const float* __restrict__ d,
                                   float* __restrict__ dmax) {
    __shared__ float red[32];
    float m = -3.4e38f;
    for (int j = threadIdx.x; j < NROWS; j += blockDim.x)
        m = fmaxf(m, d[j]);
    #pragma unroll
    for (int off = 16; off > 0; off >>= 1)
        m = fmaxf(m, __shfl_xor(m, off, 32));
    int lane = threadIdx.x & 31, wave = threadIdx.x >> 5;
    if (lane == 0) red[wave] = m;
    __syncthreads();
    if (wave == 0) {
        int nw = blockDim.x >> 5;
        float mm = (lane < nw) ? red[lane] : -3.4e38f;
        #pragma unroll
        for (int off = 16; off > 0; off >>= 1)
            mm = fmaxf(mm, __shfl_xor(mm, off, 32));
        if (lane == 0) *dmax = mm;
    }
}

// ---- kernel 4: E[j]=exp(d[j]-Dmax), F[j]=exp(0.01(d[j]-Dmax)), both <= 1 ----
__global__ void prep_ef_kernel(const float* __restrict__ d,
                               const float* __restrict__ dmax,
                               float* __restrict__ E, float* __restrict__ F) {
    int j = blockIdx.x * blockDim.x + threadIdx.x;
    float t = d[j] - *dmax;
    E[j] = __expf(t);
    F[j] = __expf(0.01f * t);
}

// ---- kernel 5: v = x @ wv_w^T + wv_b, stored TRANSPOSED bf16: vT[col][row] ----
// One wave per 16x16 output tile; bf16 WMMA over K=512 in chunks of 32.
__global__ void v_gemm_kernel(const float* __restrict__ x,
                              const float* __restrict__ wv_w,
                              const float* __restrict__ wv_b,
                              __bf16* __restrict__ vT) {
    int lane  = threadIdx.x & 31;
    int wave  = threadIdx.x >> 5;
    int l15   = lane & 15;
    int kg    = lane >> 4;          // K-group: 0 -> lanes 0-15, 1 -> lanes 16-31
    int mtile = blockIdx.x;                      // rows [mtile*16, +16)
    int ntile = blockIdx.y * 8 + wave;           // cols [ntile*16, +16)
    int row   = mtile * 16 + l15;
    int col   = ntile * 16 + l15;

    v8f acc = {};
    for (int kb = 0; kb < KDIM; kb += 32) {
        // A: x tile 16x32, contiguous K layout (lane=row, K=kg*16+e)
        const float* ap = x + (size_t)row * KDIM + kb + kg * 16;
        float af[16];
        *(float4*)(af + 0)  = *(const float4*)(ap + 0);
        *(float4*)(af + 4)  = *(const float4*)(ap + 4);
        *(float4*)(af + 8)  = *(const float4*)(ap + 8);
        *(float4*)(af + 12) = *(const float4*)(ap + 12);
        // B: wv_w^T tile 32x16 -> B[k][n] = wv_w[col][k]; lane=col, K=kg*16+e
        const float* bp = wv_w + (size_t)col * KDIM + kb + kg * 16;
        float bf[16];
        *(float4*)(bf + 0)  = *(const float4*)(bp + 0);
        *(float4*)(bf + 4)  = *(const float4*)(bp + 4);
        *(float4*)(bf + 8)  = *(const float4*)(bp + 8);
        *(float4*)(bf + 12) = *(const float4*)(bp + 12);
        v16bf A, B;
        #pragma unroll
        for (int e = 0; e < 16; ++e) {
            A[e] = (__bf16)af[e];
            B[e] = (__bf16)bf[e];
        }
        acc = __builtin_amdgcn_wmma_f32_16x16x32_bf16(
            false, A, false, B, (short)0, acc, false, false);
    }
    float bias = wv_b[col];
    // D layout: vgpr r -> M = r + kg*8 for this lane; rows contiguous -> 16B store
    v8bf o;
    #pragma unroll
    for (int r = 0; r < 8; ++r)
        o[r] = (__bf16)(acc[r] + bias);
    *(v8bf*)(vT + (size_t)col * NROWS + mtile * 16 + kg * 8) = o;
}

// ---- kernel 6: out = softmax(leaky(s_i+d_j)) @ v   (streamed, WMMA bf16) ----
// Block = 64 rows x 256 cols. wave: msub=wave>>1 (row tile), cgrp=wave&1
// (128-col slab = 8 WMMA acc tiles). Inner: j in chunks of 32 (WMMA K).
// Pure-VALU weight construction (no exp) + 9th WMMA vs ones for row sums.
__global__ void attn_kernel(const float* __restrict__ s,
                            const float* __restrict__ d,
                            const float* __restrict__ dmax,
                            const float* __restrict__ E,
                            const float* __restrict__ F,
                            const __bf16* __restrict__ vT,
                            float* __restrict__ out) {
    int lane  = threadIdx.x & 31;
    int wave  = threadIdx.x >> 5;
    int l15   = lane & 15;
    int kg    = lane >> 4;
    int msub  = wave >> 1;                       // 0..3
    int cgrp  = wave & 1;                        // 0..1
    int rbase = blockIdx.x * 64 + msub * 16;     // 16-row tile base
    int cbase = blockIdx.y * 256 + cgrp * 128;   // 128-col slab base
    int i     = rbase + l15;                     // my A-matrix row

    float Dm = *dmax;
    float si = s[i];
    float mi = leaky01(si + Dm);                 // exact row max of logits
    float P  = __expf(si + Dm - mi);             // <= 1
    float Q  = __expf(0.01f * (si + Dm) - mi);   // <= 1
    float negs = -si;

    v16bf Bones;
    #pragma unroll
    for (int e = 0; e < 16; ++e) Bones[e] = (__bf16)1.0f;

    v8f acc[8] = {};
    v8f accz = {};

    for (int jb = 0; jb < NROWS; jb += 32) {
        // ---- build A = w[i][jb..jb+32) in bf16 (contiguous K: j = jb+kg*16+e) ----
        int p = jb + kg * 16;
        float df[16], ef[16], ff[16];
        *(float4*)(df + 0)  = *(const float4*)(d + p);
        *(float4*)(df + 4)  = *(const float4*)(d + p + 4);
        *(float4*)(df + 8)  = *(const float4*)(d + p + 8);
        *(float4*)(df + 12) = *(const float4*)(d + p + 12);
        *(float4*)(ef + 0)  = *(const float4*)(E + p);
        *(float4*)(ef + 4)  = *(const float4*)(E + p + 4);
        *(float4*)(ef + 8)  = *(const float4*)(E + p + 8);
        *(float4*)(ef + 12) = *(const float4*)(E + p + 12);
        *(float4*)(ff + 0)  = *(const float4*)(F + p);
        *(float4*)(ff + 4)  = *(const float4*)(F + p + 4);
        *(float4*)(ff + 8)  = *(const float4*)(F + p + 8);
        *(float4*)(ff + 12) = *(const float4*)(F + p + 12);
        v16bf A;
        #pragma unroll
        for (int e = 0; e < 16; ++e) {
            float w = (df[e] >= negs) ? P * ef[e] : Q * ff[e];
            A[e] = (__bf16)w;
        }
        // ---- 8 col WMMAs + 1 ones-WMMA for the row sums ----
        #pragma unroll
        for (int nt = 0; nt < 8; ++nt) {
            const __bf16* bp =
                vT + (size_t)(cbase + nt * 16 + l15) * NROWS + jb + kg * 16;
            v16bf B = *(const v16bf*)bp;
            acc[nt] = __builtin_amdgcn_wmma_f32_16x16x32_bf16(
                false, A, false, B, (short)0, acc[nt], false, false);
        }
        accz = __builtin_amdgcn_wmma_f32_16x16x32_bf16(
            false, A, false, Bones, (short)0, accz, false, false);
    }

    // D layout: vgpr r -> M = r + kg*8, N = l15; Z[M] sits in accz[r] already
    #pragma unroll
    for (int r = 0; r < 8; ++r) {
        int M = r + kg * 8;
        float inv = 1.0f / accz[r];
        #pragma unroll
        for (int nt = 0; nt < 8; ++nt) {
            out[(size_t)(rbase + M) * DVC + cbase + nt * 16 + l15] =
                acc[nt][r] * inv;
        }
    }
}

// ---------------------------------------------------------------------------
extern "C" void kernel_launch(void* const* d_in, const int* in_sizes, int n_in,
                              void* d_out, int out_size, void* d_ws, size_t ws_size,
                              hipStream_t stream) {
    (void)in_sizes; (void)n_in; (void)out_size; (void)ws_size;
    const float* x    = (const float*)d_in[0];
    const float* w_w  = (const float*)d_in[1];
    const float* w_b  = (const float*)d_in[2];
    const float* wv_w = (const float*)d_in[3];
    const float* wv_b = (const float*)d_in[4];
    const float* a    = (const float*)d_in[5];
    float* out = (float*)d_out;

    float* ws    = (float*)d_ws;
    float* u1    = ws;                     // 512
    float* u2    = ws + 512;               // 512
    float* cpair = ws + 1024;              // 2
    float* dmax  = ws + 1026;              // 1
    float* sv    = ws + 2048;              // 8192
    float* dv    = sv + NROWS;             // 8192
    float* Ev    = dv + NROWS;             // 8192
    float* Fv    = Ev + NROWS;             // 8192
    __bf16* vT   = (__bf16*)(Fv + NROWS);  // 512*8192 bf16 (8 MB)

    prep_u_kernel<<<1, 512, 0, stream>>>(w_w, w_b, a, u1, u2, cpair);
    compute_sd_kernel<<<NROWS / 8, 256, 0, stream>>>(x, u1, u2, cpair, sv, dv);
    reduce_dmax_kernel<<<1, 1024, 0, stream>>>(dv, dmax);
    prep_ef_kernel<<<NROWS / 256, 256, 0, stream>>>(dv, dmax, Ev, Fv);
    v_gemm_kernel<<<dim3(NROWS / 16, DVC / 128), 256, 0, stream>>>(x, wv_w, wv_b, vT);
    attn_kernel<<<dim3(NROWS / 64, 2), 256, 0, stream>>>(sv, dv, dmax, Ev, Fv, vT, out);
}
